// SlidingWindowMultiheadAttention_24919400251442
// MI455X (gfx1250) — matrix operand order repaired and checked by reading, not compiled, not run on hardware
//
#include <hip/hip_runtime.h>

// ---------------- constants ----------------
#define B_   2
#define T_   2048
#define C_   1024
#define H_   16
#define HD_  64
#define KP   40    // padded LDS stride (bf16 elems) for 32-wide K tiles
#define KP2  72    // padded LDS stride (bf16 elems) for 64-wide tiles

typedef __bf16 bf16;
typedef __attribute__((ext_vector_type(16))) __bf16 bf16x16;
typedef __attribute__((ext_vector_type(8)))  __bf16 bf16x8;
typedef __attribute__((ext_vector_type(8)))  float  floatx8;
typedef __attribute__((ext_vector_type(4)))  unsigned int u32x4;
typedef __attribute__((ext_vector_type(8)))  int  i32x8;
typedef __attribute__((ext_vector_type(4)))  int  i32x4;

#if __has_builtin(__builtin_amdgcn_tensor_load_to_lds) && \
    __has_builtin(__builtin_amdgcn_s_wait_tensorcnt)
#define USE_TDM 1
#else
#define USE_TDM 0
#endif

union BF16x16 { bf16x16 v; bf16x8 h[2]; };

// D = A*B + C  (16x16x32 bf16 -> f32)
__device__ __forceinline__ floatx8 wmma_bf16(bf16x16 a, bf16x16 b, floatx8 c) {
  return __builtin_amdgcn_wmma_f32_16x16x32_bf16(false, a, false, b, (short)0, c,
                                                 false, false);
}

// A-matrix fragment (16x32 bf16) from row-major [m][k] source.
// Lane l<16: row=l, k = {0..7, 16..23}; lane>=16: row=l-16, k = {8..15, 24..31}.
__device__ __forceinline__ bf16x16 load_frag_a(const bf16* base, int stride, int lane) {
  const int hf = lane >> 4, lr = lane & 15;
  const bf16* p = base + lr * stride + hf * 8;
  BF16x16 u;
  u.h[0] = *(const bf16x8*)(p);
  u.h[1] = *(const bf16x8*)(p + 16);
  return u.v;
}

// Same A fragment but source is fp32 (convert to bf16 in-register).
__device__ __forceinline__ bf16x16 load_frag_a_f32(const float* base, int stride, int lane) {
  const int hf = lane >> 4, lr = lane & 15;
  const float* p = base + lr * stride + hf * 8;
  BF16x16 u;
#pragma unroll
  for (int e = 0; e < 8; ++e) {
    u.h[0][e] = (bf16)p[e];
    u.h[1][e] = (bf16)p[16 + e];
  }
  return u.v;
}

// B-matrix fragment (32x16 bf16) from [n][k] source (n = output column).
// Lane l<16: col=l, k = {0..7, 16..23}; lane>=16: col=l-16, k = {8..15, 24..31}.
__device__ __forceinline__ bf16x16 load_frag_b(const bf16* base, int stride, int lane) {
  const int hf = lane >> 4, lr = lane & 15;
  const bf16* p = base + lr * stride + hf * 8;
  BF16x16 u;
  u.h[0] = *(const bf16x8*)(p);
  u.h[1] = *(const bf16x8*)(p + 16);
  return u.v;
}

__device__ __forceinline__ float rmax16(float v) {
  v = fmaxf(v, __shfl_xor(v, 1));
  v = fmaxf(v, __shfl_xor(v, 2));
  v = fmaxf(v, __shfl_xor(v, 4));
  v = fmaxf(v, __shfl_xor(v, 8));
  return v;
}
__device__ __forceinline__ float rsum16(float v) {
  v += __shfl_xor(v, 1);
  v += __shfl_xor(v, 2);
  v += __shfl_xor(v, 4);
  v += __shfl_xor(v, 8);
  return v;
}

#if USE_TDM
// Issue one TDM 2D tile load: 64 rows x 64 bf16 (row stride HD_ elems) from
// global into LDS, with hardware row padding 128B + 16B (matches KP2 = 72).
// D# layout per CDNA5 ISA ch.8 (group0: addr/type, group1: dims/strides/pad).
__device__ __forceinline__ void tdm_load_tile64(unsigned lds_byte, const bf16* gptr) {
  const unsigned long long ga = (unsigned long long)gptr;
  u32x4 g0;
  g0[0] = 1u;                                            // count=1, user mode
  g0[1] = lds_byte;                                      // lds_addr
  g0[2] = (unsigned)ga;                                  // global_addr[31:0]
  g0[3] = (unsigned)((ga >> 32) & 0x01FFFFFFu) | (2u << 30);  // addr[56:32] | type=2
  i32x8 g1;
  g1[0] = (1 << 16) | (1 << 20) | (4 << 22) | (3 << 25); // data_size=2B, pad_en,
                                                         // pad_interval=128B, pad=4 dw
  g1[1] = (int)(64u << 16);        // tensor_dim0 = 64 (d)
  g1[2] = (int)(2048u << 16);      // tensor_dim1 = 2048 (keys)
  g1[3] = (int)(64u << 16);        // tile_dim0 = 64
  g1[4] = 64;                      // tile_dim1 = 64, tile_dim2 = 0
  g1[5] = 64;                      // tensor_dim0_stride = 64 elems
  g1[6] = 0;
  g1[7] = 0;
  const i32x4 z4 = {};
  const i32x8 z8 = {};
  // 6-arg form (clang-23 / therock headers): groups 2,3 disabled, group "4" zero.
  __builtin_amdgcn_tensor_load_to_lds(g0, g1, z4, z4, z8, 0);
}
#endif

// ---------------------------------------------------------------------------
// Kernel 1: one projection of QKV.  Y(r,j) = sum_k X(r,k) W(j,k) + bias(j)
// X: fp32 [B*T, C], W: fp32 [C, C] slice, out: bf16 in [b][head][t][hd] layout.
// Grid (32,16), block 256 (8 waves).  Tile: M=128, N=64, K-step=32.
// ---------------------------------------------------------------------------
__global__ __launch_bounds__(256)
void swmha_qkv_proj(const float* __restrict__ X, const float* __restrict__ W,
                    const float* __restrict__ bias, bf16* __restrict__ out) {
  __shared__ bf16 Ws[64 * KP];
  const int tid  = threadIdx.x;
  const int wave = tid >> 5;
  const int lane = tid & 31;
  const int m0 = blockIdx.x * 128;
  const int n0 = blockIdx.y * 64;

  floatx8 acc[4] = {};

  for (int k0 = 0; k0 < C_; k0 += 32) {
    __syncthreads();
    // Stage W tile (64 x 32) fp32 -> bf16 into LDS (shared by all 8 waves).
#pragma unroll
    for (int it = 0; it < 2; ++it) {
      const int idx = tid + it * 256;      // 0..511
      const int row = idx >> 3;            // 0..63
      const int cg  = (idx & 7) * 4;       // 0..28
      const float4 wv = *(const float4*)(W + (size_t)(n0 + row) * C_ + k0 + cg);
      bf16* dp = &Ws[row * KP + cg];
      dp[0] = (bf16)wv.x; dp[1] = (bf16)wv.y; dp[2] = (bf16)wv.z; dp[3] = (bf16)wv.w;
    }
    __syncthreads();

    const bf16x16 a =
        load_frag_a_f32(X + (size_t)(m0 + wave * 16) * C_ + k0, C_, lane);
#pragma unroll
    for (int nt = 0; nt < 4; ++nt) {
      const bf16x16 bfr = load_frag_b(&Ws[(nt * 16) * KP], KP, lane);
      acc[nt] = wmma_bf16(a, bfr, acc[nt]);
    }
  }

  // Epilogue: bias add + scatter into [b][head][t][hd] bf16.
  const int hf = lane >> 4, lr = lane & 15;
#pragma unroll
  for (int nt = 0; nt < 4; ++nt) {
    const int n = n0 + nt * 16 + lr;            // 0..1023 within this projection
    const float bv = bias[n];
    const int head = n >> 6, dd = n & 63;
#pragma unroll
    for (int i = 0; i < 8; ++i) {
      const int m  = m0 + wave * 16 + hf * 8 + i;   // 0..4095
      const int bi = m >> 11;
      const int ti = m & (T_ - 1);
      out[(((size_t)bi * H_ + head) * T_ + ti) * HD_ + dd] = (bf16)(acc[nt][i] + bv);
    }
  }
}

// ---------------------------------------------------------------------------
// Kernel 2: flash attention with +1 band bias inside +/-window.
// Q,K,V: bf16 [b*h][t][64]. Aout: bf16 [b][t][c].
// Grid (T/64, B*H), block 128 (4 waves, 16 q-rows per wave).
// K tile and V tile are DMA'd into LDS by the Tensor Data Mover (waves 0/1
// drive two concurrent descriptors); V is then transposed LDS->LDS for the
// B-fragment layout of the PV matmul.
// ---------------------------------------------------------------------------
__global__ __launch_bounds__(128)
void swmha_flash_attn(const bf16* __restrict__ Q, const bf16* __restrict__ K,
                      const bf16* __restrict__ V, bf16* __restrict__ Aout,
                      const int* __restrict__ win_p) {
  __shared__ bf16 Ks[64 * KP2];       // [key][d]
#if USE_TDM
  __shared__ bf16 Vs[64 * KP2];       // [key][d] (TDM staging for V)
#endif
  __shared__ bf16 Vt[64 * KP2];       // [d][key] (transposed)
  __shared__ bf16 Ps[4][16 * KP2];    // per-wave P scratch [row][key]

  const int tid  = threadIdx.x;
  const int wave = tid >> 5;
  const int lane = tid & 31;
  const int hf = lane >> 4, lr = lane & 15;
  const int bh = blockIdx.y;
  const int q0 = blockIdx.x * 64 + wave * 16;
  const int win = win_p[0];
  const float scale = 0.125f;  // 1/sqrt(64)

  const bf16* Qb = Q + (size_t)bh * T_ * HD_;
  const bf16* Kb = K + (size_t)bh * T_ * HD_;
  const bf16* Vb = V + (size_t)bh * T_ * HD_;

  // Q fragments stay in registers for the whole kernel (16 rows x 64 d).
  const bf16x16 aq0 = load_frag_a(Qb + (size_t)q0 * HD_ + 0, HD_, lane);
  const bf16x16 aq1 = load_frag_a(Qb + (size_t)q0 * HD_ + 32, HD_, lane);

  floatx8 o[4] = {};
  float rowmax[8], rowsum[8];
#pragma unroll
  for (int i = 0; i < 8; ++i) { rowmax[i] = -1e30f; rowsum[i] = 0.0f; }

  bf16* Pw = &Ps[wave][0];

  for (int kt = 0; kt < T_ / 64; ++kt) {
    const int kb = kt * 64;
    __syncthreads();   // previous iteration's LDS reads complete

#if USE_TDM
    // Tensor Data Mover: two concurrent 2D descriptors (K by wave 0, V by
    // wave 1), hardware-padded to the KP2 LDS stride.  TENSORcnt tracked.
    if (wave == 0) {
      tdm_load_tile64((unsigned)(unsigned long long)(void*)&Ks[0],
                      Kb + (size_t)kb * HD_);
      __builtin_amdgcn_s_wait_tensorcnt(0);
    } else if (wave == 1) {
      tdm_load_tile64((unsigned)(unsigned long long)(void*)&Vs[0],
                      Vb + (size_t)kb * HD_);
      __builtin_amdgcn_s_wait_tensorcnt(0);
    }
    __syncthreads();   // Ks/Vs visible to all waves
    // Transpose V tile LDS->LDS for the PV B-fragments.
#pragma unroll
    for (int it = 0; it < 4; ++it) {
      const int idx = tid + it * 128;   // 0..511
      const int row = idx >> 3;         // key 0..63
      const int dg  = (idx & 7) * 8;    // d group
      const bf16x8 vv = *(const bf16x8*)&Vs[row * KP2 + dg];
#pragma unroll
      for (int e = 0; e < 8; ++e) Vt[(dg + e) * KP2 + row] = vv[e];
    }
#else
    // Fallback: VALU staging of K straight and V transposed.
#pragma unroll
    for (int it = 0; it < 4; ++it) {
      const int idx = tid + it * 128;
      const int row = idx >> 3;
      const int dg  = (idx & 7) * 8;
      const bf16x8 kv = *(const bf16x8*)(Kb + (size_t)(kb + row) * HD_ + dg);
      *(bf16x8*)&Ks[row * KP2 + dg] = kv;
      const bf16x8 vv = *(const bf16x8*)(Vb + (size_t)(kb + row) * HD_ + dg);
#pragma unroll
      for (int e = 0; e < 8; ++e) Vt[(dg + e) * KP2 + row] = vv[e];
    }
#endif
    __syncthreads();

    if (kt + 1 < T_ / 64)
      __builtin_prefetch((const void*)(Kb + (size_t)(kb + 64) * HD_), 0, 3);

    // S = Q K^T  (16 rows x 64 keys as 4 C-tiles).
    floatx8 s[4];
#pragma unroll
    for (int nt = 0; nt < 4; ++nt) {
      const bf16x16 bk0 = load_frag_b(&Ks[(nt * 16) * KP2 + 0],  KP2, lane);
      const bf16x16 bk1 = load_frag_b(&Ks[(nt * 16) * KP2 + 32], KP2, lane);
      floatx8 z = {};
      z = wmma_bf16(aq0, bk0, z);
      z = wmma_bf16(aq1, bk1, z);
      s[nt] = z;
    }

    // Logits: scale + band bias; online softmax update.
    float pbuf[4][8], nm[8];
#pragma unroll
    for (int i = 0; i < 8; ++i) nm[i] = rowmax[i];
#pragma unroll
    for (int nt = 0; nt < 4; ++nt) {
      const int kc = kb + nt * 16 + lr;
#pragma unroll
      for (int i = 0; i < 8; ++i) {
        const int qr = q0 + hf * 8 + i;
        const int d = qr - kc;
        const float band = ((d < 0 ? -d : d) <= win) ? 1.0f : 0.0f;
        const float lg = s[nt][i] * scale + band;
        pbuf[nt][i] = lg;
        nm[i] = fmaxf(nm[i], lg);
      }
    }
#pragma unroll
    for (int i = 0; i < 8; ++i) nm[i] = rmax16(nm[i]);

    float corr[8], lsum[8];
#pragma unroll
    for (int i = 0; i < 8; ++i) {
      corr[i] = __expf(rowmax[i] - nm[i]);
      rowmax[i] = nm[i];
      lsum[i] = 0.0f;
    }
#pragma unroll
    for (int nt = 0; nt < 4; ++nt)
#pragma unroll
      for (int i = 0; i < 8; ++i) {
        const float pv = __expf(pbuf[nt][i] - nm[i]);
        pbuf[nt][i] = pv;
        lsum[i] += pv;
      }
#pragma unroll
    for (int i = 0; i < 8; ++i)
      rowsum[i] = rowsum[i] * corr[i] + rsum16(lsum[i]);
#pragma unroll
    for (int dt = 0; dt < 4; ++dt)
#pragma unroll
      for (int i = 0; i < 8; ++i) o[dt][i] *= corr[i];

    // P: C-layout -> LDS -> A-layout (per-wave region; same-wave DS ordering).
#pragma unroll
    for (int nt = 0; nt < 4; ++nt)
#pragma unroll
      for (int i = 0; i < 8; ++i)
        Pw[(hf * 8 + i) * KP2 + nt * 16 + lr] = (bf16)pbuf[nt][i];

    const bf16x16 ap0 = load_frag_a(Pw + 0,  KP2, lane);
    const bf16x16 ap1 = load_frag_a(Pw + 32, KP2, lane);

    // O += P V   (B fragments from transposed V: [d][key]).
#pragma unroll
    for (int dt = 0; dt < 4; ++dt) {
      const bf16x16 bv0 = load_frag_b(&Vt[(dt * 16) * KP2 + 0],  KP2, lane);
      const bf16x16 bv1 = load_frag_b(&Vt[(dt * 16) * KP2 + 32], KP2, lane);
      o[dt] = wmma_bf16(ap0, bv0, o[dt]);
      o[dt] = wmma_bf16(ap1, bv1, o[dt]);
    }
  }

  // Normalize and write attention output back in [b][t][c] bf16.
  float inv[8];
#pragma unroll
  for (int i = 0; i < 8; ++i) inv[i] = 1.0f / rowsum[i];
  const int bi = bh >> 4, head = bh & 15;
#pragma unroll
  for (int dt = 0; dt < 4; ++dt) {
    const int col = head * 64 + dt * 16 + lr;
#pragma unroll
    for (int i = 0; i < 8; ++i) {
      const int ti = q0 + hf * 8 + i;
      Aout[((size_t)bi * T_ + ti) * C_ + col] = (bf16)(o[dt][i] * inv[i]);
    }
  }
}

// ---------------------------------------------------------------------------
// Kernel 3: out projection.  Y = A(bf16) * Wo^T + bo  -> fp32 d_out.
// Grid (32,16), block 256.
// ---------------------------------------------------------------------------
__global__ __launch_bounds__(256)
void swmha_out_proj(const bf16* __restrict__ A, const float* __restrict__ W,
                    const float* __restrict__ bias, float* __restrict__ Y) {
  __shared__ bf16 Ws[64 * KP];
  const int tid  = threadIdx.x;
  const int wave = tid >> 5;
  const int lane = tid & 31;
  const int m0 = blockIdx.x * 128;
  const int n0 = blockIdx.y * 64;

  floatx8 acc[4] = {};

  for (int k0 = 0; k0 < C_; k0 += 32) {
    __syncthreads();
#pragma unroll
    for (int it = 0; it < 2; ++it) {
      const int idx = tid + it * 256;
      const int row = idx >> 3;
      const int cg  = (idx & 7) * 4;
      const float4 wv = *(const float4*)(W + (size_t)(n0 + row) * C_ + k0 + cg);
      bf16* dp = &Ws[row * KP + cg];
      dp[0] = (bf16)wv.x; dp[1] = (bf16)wv.y; dp[2] = (bf16)wv.z; dp[3] = (bf16)wv.w;
    }
    __syncthreads();

    const bf16x16 a =
        load_frag_a(A + (size_t)(m0 + wave * 16) * C_ + k0, C_, lane);
#pragma unroll
    for (int nt = 0; nt < 4; ++nt) {
      const bf16x16 bfr = load_frag_b(&Ws[(nt * 16) * KP], KP, lane);
      acc[nt] = wmma_bf16(a, bfr, acc[nt]);
    }
  }

  const int hf = lane >> 4, lr = lane & 15;
#pragma unroll
  for (int nt = 0; nt < 4; ++nt) {
    const int n = n0 + nt * 16 + lr;
    const float bv = bias[n];
#pragma unroll
    for (int i = 0; i < 8; ++i) {
      const int m = m0 + wave * 16 + hf * 8 + i;
      Y[(size_t)m * C_ + n] = acc[nt][i] + bv;
    }
  }
}

// ---------------------------------------------------------------------------
extern "C" void kernel_launch(void* const* d_in, const int* in_sizes, int n_in,
                              void* d_out, int out_size, void* d_ws, size_t ws_size,
                              hipStream_t stream) {
  const float* query = (const float*)d_in[0];
  const float* key   = (const float*)d_in[1];
  const float* value = (const float*)d_in[2];
  const float* w_in  = (const float*)d_in[3];   // [3C, C]
  const float* b_in  = (const float*)d_in[4];   // [3C]
  const float* w_out = (const float*)d_in[5];   // [C, C]
  const float* b_out = (const float*)d_in[6];   // [C]
  const int*   winp  = (const int*)d_in[7];
  float* out = (float*)d_out;

  const size_t NT = (size_t)B_ * H_ * T_ * HD_;   // 4 Mi elements per tensor
  bf16* q_ws = (bf16*)d_ws;
  bf16* k_ws = q_ws + NT;
  bf16* v_ws = k_ws + NT;
  bf16* a_ws = v_ws + NT;                          // total: 32 MB bf16

  const dim3 gp(32, 16, 1), bp(256, 1, 1);
  swmha_qkv_proj<<<gp, bp, 0, stream>>>(query, w_in,               b_in,          q_ws);
  swmha_qkv_proj<<<gp, bp, 0, stream>>>(key,   w_in + (size_t)C_ * C_,     b_in + C_,     k_ws);
  swmha_qkv_proj<<<gp, bp, 0, stream>>>(value, w_in + (size_t)2 * C_ * C_, b_in + 2 * C_, v_ws);

  swmha_flash_attn<<<dim3(T_ / 64, B_ * H_, 1), dim3(128, 1, 1), 0, stream>>>(
      q_ws, k_ws, v_ws, a_ws, winp);

  swmha_out_proj<<<gp, bp, 0, stream>>>(a_ws, w_out, b_out, out);
}
